// BaseH2XAttLayer_2259152797788
// MI455X (gfx1250) — matrix operand order repaired and checked by reading, not compiled, not run on hardware
//
#include <hip/hip_runtime.h>
#include <hip/hip_bf16.h>

typedef __attribute__((ext_vector_type(16))) _Float16 v16h;
typedef __attribute__((ext_vector_type(8)))  float    v8f;

#define IN_DIM  128
#define NH      16
#define HD      8
#define EDGE_F  4
#define R_F     20
#define KV_IN   280
#define KV_PAD  288
#define EPB     32            // edges per block in edge_kernel
#define LN_EPS  1e-5f
#define INV_SQRT_HD 0.35355339059327373f

// ---------------- WMMA fragment loaders (wave32, 16x16x32 f16) ----------------
// A: 16x32 f16. lanes 0-15: row M, K in {0..7,16..23}; lanes 16-31: K in {8..15,24..31}.
__device__ inline v16h load_a_frag(const _Float16* A, int lda, int k0) {
  const int lane = threadIdx.x & 31;
  const int r  = lane & 15;
  const int kb = (lane & 16) ? 8 : 0;
  v16h a;
#pragma unroll
  for (int v = 0; v < 8; ++v) {
    int k = k0 + kb + ((v & 3) << 1) + ((v & 4) ? 16 : 0);
    a[2 * v]     = A[r * lda + k];
    a[2 * v + 1] = A[r * lda + k + 1];
  }
  return a;
}

// B: 32x16 f16, weights stored transposed as Wt[col * kpad + k].
__device__ inline v16h load_b_frag(const _Float16* Wt, int kpad, int col0, int k0) {
  const int lane = threadIdx.x & 31;
  const int c  = col0 + (lane & 15);
  const int kb = (lane & 16) ? 8 : 0;
  v16h b;
#pragma unroll
  for (int v = 0; v < 8; ++v) {
    int k = k0 + kb + ((v & 3) << 1) + ((v & 4) ? 16 : 0);
    b[2 * v]     = Wt[c * kpad + k];
    b[2 * v + 1] = Wt[c * kpad + k + 1];
  }
  return b;
}

// monotonic float <-> uint for atomicMax over signed floats
__device__ inline unsigned f2key(float f) {
  unsigned u = __float_as_uint(f);
  return (u & 0x80000000u) ? ~u : (u | 0x80000000u);
}
__device__ inline float key2f(unsigned k) {
  unsigned u = (k & 0x80000000u) ? (k & 0x7FFFFFFFu) : ~k;
  return __uint_as_float(u);
}

// ---------------- weight prep: f32 [K x N] -> f16 transposed+padded [N x Kpad] --
__global__ void wtrans_kernel(const float* __restrict__ src, _Float16* __restrict__ dst,
                              int K, int Nc, int Kpad) {
  int idx = blockIdx.x * blockDim.x + threadIdx.x;
  int total = Nc * Kpad;
  if (idx >= total) return;
  int n = idx / Kpad, k = idx - n * Kpad;
  float v = (k < K) ? src[k * Nc + n] : 0.f;
  dst[idx] = (_Float16)v;
}

// ---------------- zero init for out / denom / segmax ----------------
__global__ void init_kernel(float* __restrict__ out, float* __restrict__ denom,
                            unsigned* __restrict__ smax, int N) {
  int i = blockIdx.x * blockDim.x + threadIdx.x;
  if (i < N * 3)  out[i]   = 0.f;
  if (i < N * NH) { denom[i] = 0.f; smax[i] = 0u; }
}

// ---------------- q MLP over nodes: h[N,128] -> q[N,128] ----------------
__global__ __launch_bounds__(256)
void q_mlp_kernel(const float* __restrict__ h,
                  const _Float16* __restrict__ W1t, const float* __restrict__ b1,
                  const float* __restrict__ gg, const float* __restrict__ be,
                  const _Float16* __restrict__ W2t, const float* __restrict__ b2,
                  float* __restrict__ qout, int N) {
  __shared__ _Float16 Ah[16 * 128];
  __shared__ float    hid[16 * 128];
  __shared__ _Float16 Ahid[16 * 128];
  __shared__ float    part[16][16][2];
  __shared__ float    stats[16][2];
  const int tid = threadIdx.x, wave = tid >> 5, lane = tid & 31;
  const int n0 = blockIdx.x * 16;

  for (int i = tid; i < 16 * 128; i += 256) {
    int r = i >> 7, c = i & 127;
    int row = n0 + r; if (row >= N) row = N - 1;
    Ah[i] = (_Float16)h[row * 128 + c];
  }
  __syncthreads();

  { // layer 1
    v8f acc = {};
#pragma unroll
    for (int kk = 0; kk < 4; ++kk) {
      v16h a = load_a_frag(Ah, 128, kk * 32);
      v16h b = load_b_frag(W1t, 128, wave * 16, kk * 32);
      acc = __builtin_amdgcn_wmma_f32_16x16x32_f16(false, a, false, b, (short)0, acc, false, false);
    }
    int col = wave * 16 + (lane & 15);
    int rb = (lane & 16) ? 8 : 0;
#pragma unroll
    for (int i = 0; i < 8; ++i) hid[(rb + i) * 128 + col] = acc[i] + b1[col];
  }
  __syncthreads();

  { // LayerNorm partials: 16 rows x 16 segments of 8
    int r = tid >> 4, seg = tid & 15;
    float s = 0.f, s2 = 0.f;
#pragma unroll
    for (int j = 0; j < 8; ++j) {
      float x = hid[r * 128 + seg * 8 + j];
      s += x; s2 += x * x;
    }
    part[r][seg][0] = s; part[r][seg][1] = s2;
  }
  __syncthreads();
  if (tid < 16) {
    float s = 0.f, s2 = 0.f;
    for (int j = 0; j < 16; ++j) { s += part[tid][j][0]; s2 += part[tid][j][1]; }
    float mu = s / 128.f;
    float var = s2 / 128.f - mu * mu;
    stats[tid][0] = mu; stats[tid][1] = rsqrtf(var + LN_EPS);
  }
  __syncthreads();
  for (int i = tid; i < 16 * 128; i += 256) {
    int r = i >> 7, c = i & 127;
    float x = (hid[i] - stats[r][0]) * stats[r][1] * gg[c] + be[c];
    Ahid[i] = (_Float16)fmaxf(x, 0.f);
  }
  __syncthreads();

  { // layer 2
    v8f acc = {};
#pragma unroll
    for (int kk = 0; kk < 4; ++kk) {
      v16h a = load_a_frag(Ahid, 128, kk * 32);
      v16h b = load_b_frag(W2t, 128, wave * 16, kk * 32);
      acc = __builtin_amdgcn_wmma_f32_16x16x32_f16(false, a, false, b, (short)0, acc, false, false);
    }
    int col = wave * 16 + (lane & 15);
    int rb = (lane & 16) ? 8 : 0;
#pragma unroll
    for (int i = 0; i < 8; ++i) {
      int row = n0 + rb + i;
      if (row < N) qout[row * 128 + col] = acc[i] + b2[col];
    }
  }
}

// ---------------- edge kernel: 32 edges/block, kv MLPs + gate + scores --------
__global__ __launch_bounds__(256)
void edge_kernel(const float* __restrict__ h,
                 const float* __restrict__ r_feat, const float* __restrict__ edge_feat,
                 const int* __restrict__ edge_index,
                 const _Float16* __restrict__ kW1t, const float* __restrict__ kb1,
                 const float* __restrict__ kg,   const float* __restrict__ kbe,
                 const _Float16* __restrict__ kW2t, const float* __restrict__ kb2,
                 const _Float16* __restrict__ vW1t, const float* __restrict__ vb1,
                 const float* __restrict__ vg,   const float* __restrict__ vbe,
                 const _Float16* __restrict__ vW2t, const float* __restrict__ vb2,
                 const float* __restrict__ ewW,  const float* __restrict__ ewb,
                 const float* __restrict__ qf,
                 float* __restrict__ scores, float* __restrict__ vout,
                 unsigned* __restrict__ smax, int E) {
  __shared__ _Float16 Akv[EPB * KV_PAD];          // 18.0 KB
  __shared__ float    hidK[EPB * 128];            // 16 KB
  __shared__ float    hidV[EPB * 128];            // 16 KB
  __shared__ _Float16 AhK[EPB * 128];             // 8 KB
  __shared__ _Float16 AhV[EPB * 128];             // 8 KB
  __shared__ float    part[64][4][2];             // 2 KB
  __shared__ float    stats[64][2];
  __shared__ float    ew[EPB];
  __shared__ int      sdst[EPB], ssrc[EPB];

  const int tid = threadIdx.x, wave = tid >> 5, lane = tid & 31;
  const int e0 = blockIdx.x * EPB;

  if (tid < EPB) {
    int e = e0 + tid; if (e >= E) e = E - 1;
    ssrc[tid] = edge_index[e];
    sdst[tid] = edge_index[E + e];
    float s = ewb[0];
#pragma unroll
    for (int i = 0; i < R_F; ++i) s += r_feat[e * R_F + i] * ewW[i];
    ew[tid] = 1.f / (1.f + expf(-s));
  }
  __syncthreads();

  // gather kv_input = [edge_feat(4) | r_feat(20) | h[dst](128) | h[src](128) | pad(8)]
  for (int i = tid; i < EPB * KV_PAD; i += 256) {
    int r = i / KV_PAD, k = i - r * KV_PAD;
    int e = e0 + r; if (e >= E) e = E - 1;
    float v;
    if      (k < EDGE_F)             v = edge_feat[e * EDGE_F + k];
    else if (k < EDGE_F + R_F)       v = r_feat[e * R_F + (k - EDGE_F)];
    else if (k < EDGE_F + R_F + 128) v = h[sdst[r] * 128 + (k - 24)];
    else if (k < KV_IN)              v = h[ssrc[r] * 128 + (k - 152)];
    else                             v = 0.f;
    Akv[i] = (_Float16)v;
  }
  __syncthreads();

  { // layer 1 for K and V MLPs over two 16-edge tiles; each B frag feeds 2 WMMAs
    v8f accK0 = {}, accK1 = {}, accV0 = {}, accV1 = {};
#pragma unroll
    for (int kk = 0; kk < 9; ++kk) {
      v16h a0 = load_a_frag(Akv, KV_PAD, kk * 32);
      v16h a1 = load_a_frag(Akv + 16 * KV_PAD, KV_PAD, kk * 32);
      v16h bk = load_b_frag(kW1t, KV_PAD, wave * 16, kk * 32);
      accK0 = __builtin_amdgcn_wmma_f32_16x16x32_f16(false, a0, false, bk, (short)0, accK0, false, false);
      accK1 = __builtin_amdgcn_wmma_f32_16x16x32_f16(false, a1, false, bk, (short)0, accK1, false, false);
      v16h bv = load_b_frag(vW1t, KV_PAD, wave * 16, kk * 32);
      accV0 = __builtin_amdgcn_wmma_f32_16x16x32_f16(false, a0, false, bv, (short)0, accV0, false, false);
      accV1 = __builtin_amdgcn_wmma_f32_16x16x32_f16(false, a1, false, bv, (short)0, accV1, false, false);
    }
    int col = wave * 16 + (lane & 15);
    int rb = (lane & 16) ? 8 : 0;
    float bK = kb1[col], bV = vb1[col];
#pragma unroll
    for (int i = 0; i < 8; ++i) {
      hidK[(rb + i) * 128 + col]        = accK0[i] + bK;
      hidK[(16 + rb + i) * 128 + col]   = accK1[i] + bK;
      hidV[(rb + i) * 128 + col]        = accV0[i] + bV;
      hidV[(16 + rb + i) * 128 + col]   = accV1[i] + bV;
    }
  }
  __syncthreads();

  { // LayerNorm partials: 64 (matrix,row) pairs x 4 segments of 32
    int p = tid >> 2, seg = tid & 3;
    const float* src = (p < EPB) ? &hidK[p * 128] : &hidV[(p - EPB) * 128];
    float s = 0.f, s2 = 0.f;
#pragma unroll
    for (int j = 0; j < 32; ++j) { float x = src[seg * 32 + j]; s += x; s2 += x * x; }
    part[p][seg][0] = s; part[p][seg][1] = s2;
  }
  __syncthreads();
  if (tid < 64) {
    float s = 0.f, s2 = 0.f;
#pragma unroll
    for (int j = 0; j < 4; ++j) { s += part[tid][j][0]; s2 += part[tid][j][1]; }
    float mu = s / 128.f;
    float var = s2 / 128.f - mu * mu;
    stats[tid][0] = mu; stats[tid][1] = rsqrtf(var + LN_EPS);
  }
  __syncthreads();
  for (int i = tid; i < EPB * 128; i += 256) {
    int r = i >> 7, c = i & 127;
    float x = (hidK[i] - stats[r][0]) * stats[r][1] * kg[c] + kbe[c];
    AhK[i] = (_Float16)fmaxf(x, 0.f);
    float y = (hidV[i] - stats[EPB + r][0]) * stats[EPB + r][1] * vg[c] + vbe[c];
    AhV[i] = (_Float16)fmaxf(y, 0.f);
  }
  __syncthreads();

  { // layer 2 of K MLP over both tiles; stage k into hidK for score dots
    v8f acc0 = {}, acc1 = {};
#pragma unroll
    for (int kk = 0; kk < 4; ++kk) {
      v16h b  = load_b_frag(kW2t, 128, wave * 16, kk * 32);
      v16h a0 = load_a_frag(AhK, 128, kk * 32);
      v16h a1 = load_a_frag(AhK + 16 * 128, 128, kk * 32);
      acc0 = __builtin_amdgcn_wmma_f32_16x16x32_f16(false, a0, false, b, (short)0, acc0, false, false);
      acc1 = __builtin_amdgcn_wmma_f32_16x16x32_f16(false, a1, false, b, (short)0, acc1, false, false);
    }
    int col = wave * 16 + (lane & 15);
    int rb = (lane & 16) ? 8 : 0;
    float bk2 = kb2[col];
#pragma unroll
    for (int i = 0; i < 8; ++i) {
      hidK[(rb + i) * 128 + col]      = acc0[i] + bk2;
      hidK[(16 + rb + i) * 128 + col] = acc1[i] + bk2;
    }
  }
  if (wave < 2) { // layer 2 of V MLP (16 output heads): wave w owns edge tile w
    v8f acc = {};
#pragma unroll
    for (int kk = 0; kk < 4; ++kk) {
      v16h a = load_a_frag(AhV + wave * 16 * 128, 128, kk * 32);
      v16h b = load_b_frag(vW2t, 128, 0, kk * 32);
      acc = __builtin_amdgcn_wmma_f32_16x16x32_f16(false, a, false, b, (short)0, acc, false, false);
    }
    int col = lane & 15;
    int rb = (lane & 16) ? 8 : 0;
#pragma unroll
    for (int i = 0; i < 8; ++i) {
      int r = wave * 16 + rb + i, ge = e0 + r;
      if (ge < E) vout[ge * NH + col] = (acc[i] + vb2[col]) * ew[r];
    }
  }
  __syncthreads();

  // scores = (q[dst] . k)/sqrt(HD) per head; segment max via atomicMax
  for (int idx = tid; idx < EPB * NH; idx += 256) {
    int e = idx >> 4, hd = idx & 15;
    int ge = e0 + e;
    if (ge < E) {
      int dst = sdst[e];
      float s = 0.f;
#pragma unroll
      for (int d = 0; d < HD; ++d)
        s += qf[dst * 128 + hd * HD + d] * hidK[e * 128 + hd * HD + d];
      s *= INV_SQRT_HD;
      scores[ge * NH + hd] = s;
      atomicMax(&smax[dst * NH + hd], f2key(s));
    }
  }
}

// ---------------- softmax pass 2: exp + denominator ----------------
__global__ void softmax_pass2(const int* __restrict__ edge_index,
                              const unsigned* __restrict__ smax,
                              float* __restrict__ scores, float* __restrict__ denom, int E) {
  int idx = blockIdx.x * blockDim.x + threadIdx.x;
  if (idx >= E * NH) return;
  int e = idx >> 4, hd = idx & 15;
  int dst = edge_index[E + e];
  float m = key2f(smax[dst * NH + hd]);
  float ex = expf(scores[idx] - m);
  scores[idx] = ex;
  atomicAdd(&denom[dst * NH + hd], ex);
}

// ---------------- pass 3: alpha * v * rel_x, segment-sum, head mean -----------
__global__ void scatter_out(const int* __restrict__ edge_index,
                            const float* __restrict__ rel_x,
                            const float* __restrict__ ex, const float* __restrict__ denom,
                            const float* __restrict__ v, float* __restrict__ out, int E) {
  int e = blockIdx.x * blockDim.x + threadIdx.x;
  if (e >= E) return;
  int dst = edge_index[E + e];
  float s = 0.f;
#pragma unroll
  for (int hd = 0; hd < NH; ++hd)
    s += ex[e * NH + hd] / denom[dst * NH + hd] * v[e * NH + hd];
  s *= (1.f / (float)NH);
#pragma unroll
  for (int c = 0; c < 3; ++c)
    atomicAdd(&out[dst * 3 + c], s * rel_x[e * 3 + c]);
}

// ---------------- launcher ----------------
extern "C" void kernel_launch(void* const* d_in, const int* in_sizes, int n_in,
                              void* d_out, int out_size, void* d_ws, size_t ws_size,
                              hipStream_t stream) {
  const float* h         = (const float*)d_in[0];
  const float* rel_x     = (const float*)d_in[1];
  const float* r_feat    = (const float*)d_in[2];
  const float* edge_feat = (const float*)d_in[3];
  const int*   edge_idx  = (const int*)d_in[4];
  const float* xk_W1 = (const float*)d_in[5];
  const float* xk_b1 = (const float*)d_in[6];
  const float* xk_g  = (const float*)d_in[7];
  const float* xk_be = (const float*)d_in[8];
  const float* xk_W2 = (const float*)d_in[9];
  const float* xk_b2 = (const float*)d_in[10];
  const float* xv_W1 = (const float*)d_in[11];
  const float* xv_b1 = (const float*)d_in[12];
  const float* xv_g  = (const float*)d_in[13];
  const float* xv_be = (const float*)d_in[14];
  const float* xv_W2 = (const float*)d_in[15];
  const float* xv_b2 = (const float*)d_in[16];
  const float* xq_W1 = (const float*)d_in[17];
  const float* xq_b1 = (const float*)d_in[18];
  const float* xq_g  = (const float*)d_in[19];
  const float* xq_be = (const float*)d_in[20];
  const float* xq_W2 = (const float*)d_in[21];
  const float* xq_b2 = (const float*)d_in[22];
  const float* ew_W  = (const float*)d_in[23];
  const float* ew_b  = (const float*)d_in[24];
  float* out = (float*)d_out;

  const int N = in_sizes[0] / IN_DIM;
  const int E = in_sizes[1] / 3;

  // workspace carve-up (256B aligned)
  char* p = (char*)d_ws;
  auto carve = [&](size_t bytes) { void* r = (void*)p; p += (bytes + 255) & ~(size_t)255; return r; };
  _Float16* kW1t = (_Float16*)carve((size_t)128 * KV_PAD * 2);
  _Float16* vW1t = (_Float16*)carve((size_t)128 * KV_PAD * 2);
  _Float16* qW1t = (_Float16*)carve((size_t)128 * 128 * 2);
  _Float16* kW2t = (_Float16*)carve((size_t)128 * 128 * 2);
  _Float16* vW2t = (_Float16*)carve((size_t)NH * 128 * 2);
  _Float16* qW2t = (_Float16*)carve((size_t)128 * 128 * 2);
  float*    qbuf   = (float*)carve((size_t)N * 128 * 4);
  float*    scores = (float*)carve((size_t)E * NH * 4);
  float*    vbuf   = (float*)carve((size_t)E * NH * 4);
  float*    denom  = (float*)carve((size_t)N * NH * 4);
  unsigned* smax   = (unsigned*)carve((size_t)N * NH * 4);

  // weight prep
  auto wlaunch = [&](const float* s, _Float16* d, int K, int Nc, int Kpad) {
    int total = Nc * Kpad;
    wtrans_kernel<<<(total + 255) / 256, 256, 0, stream>>>(s, d, K, Nc, Kpad);
  };
  wlaunch(xk_W1, kW1t, KV_IN, 128, KV_PAD);
  wlaunch(xv_W1, vW1t, KV_IN, 128, KV_PAD);
  wlaunch(xq_W1, qW1t, 128, 128, 128);
  wlaunch(xk_W2, kW2t, 128, 128, 128);
  wlaunch(xv_W2, vW2t, 128, NH, 128);
  wlaunch(xq_W2, qW2t, 128, 128, 128);

  init_kernel<<<(N * NH + 255) / 256, 256, 0, stream>>>(out, denom, smax, N);

  q_mlp_kernel<<<(N + 15) / 16, 256, 0, stream>>>(
      h, qW1t, xq_b1, xq_g, xq_be, qW2t, xq_b2, qbuf, N);

  edge_kernel<<<(E + EPB - 1) / EPB, 256, 0, stream>>>(
      h, r_feat, edge_feat, edge_idx,
      kW1t, xk_b1, xk_g, xk_be, kW2t, xk_b2,
      vW1t, xv_b1, xv_g, xv_be, vW2t, xv_b2,
      ew_W, ew_b, qbuf, scores, vbuf, smax, E);

  softmax_pass2<<<(E * NH + 255) / 256, 256, 0, stream>>>(edge_idx, smax, scores, denom, E);

  scatter_out<<<(E + 255) / 256, 256, 0, stream>>>(edge_idx, rel_x, scores, denom, vbuf, out, E);
}
